// Cross_58677843198640
// MI455X (gfx1250) — matrix operand order repaired and checked by reading, not compiled, not run on hardware
//
#include <hip/hip_runtime.h>
#include <hip/hip_bf16.h>

typedef __attribute__((ext_vector_type(16))) _Float16 v16h;
typedef __attribute__((ext_vector_type(8)))  float    v8f;

#define D_CH   64
#define KDIM   128
#define SDIM   128
#define M_B    2
#define NSEQ   256            // M * S  (or M * K)
#define LSEQ   128
#define NL     (NSEQ * LSEQ)  // 32768
#define TEN_ELEMS (M_B * D_CH * KDIM * SDIM)   // 2097152 per tensor

// ---------------------------------------------------------------------------
// WMMA fragment loaders (wave32, 16-bit A 16x32, B 32x16), contiguous k = 16*hi + e in A and B
// ---------------------------------------------------------------------------
__device__ __forceinline__ int a_kidx(int e, int hi) {
    return hi * 16 + e;
}

// A fragment: rows row0..row0+15, K = k0..k0+31, src row-major stride `ld`
__device__ __forceinline__ v16h load_A(const _Float16* src, int ld, int row0, int k0,
                                       int ln, int hi) {
    v16h a;
#pragma unroll
    for (int e = 0; e < 16; ++e)
        a[e] = src[(row0 + ln) * ld + k0 + a_kidx(e, hi)];
    return a;
}

// B fragment where B(k,n) = S[n][k]  (source row-major [n][k], stride ld)
__device__ __forceinline__ v16h load_B_nk(const _Float16* src, int ld, int n0, int k0,
                                          int ln, int hi) {
    v16h b;
#pragma unroll
    for (int e = 0; e < 16; ++e)
        b[e] = src[(n0 + ln) * ld + k0 + hi * 16 + e];
    return b;
}

// B fragment where B(k,n) = S[k][n]  (source row-major [k][n], stride ld)
__device__ __forceinline__ v16h load_B_kn(const _Float16* src, int ld, int k0, int n0,
                                          int ln, int hi) {
    v16h b;
#pragma unroll
    for (int e = 0; e < 16; ++e)
        b[e] = src[(k0 + hi * 16 + e) * ld + n0 + ln];
    return b;
}

// ---------------------------------------------------------------------------
// BatchNorm statistics: per channel d over all (m,k,s); folded scale/shift
// ---------------------------------------------------------------------------
__global__ __launch_bounds__(256)
void bn_stats_kernel(const float* __restrict__ xs, const float* __restrict__ xn,
                     const float* __restrict__ bn_w, const float* __restrict__ bn_b,
                     float* __restrict__ scale, float* __restrict__ shift) {
    int t = blockIdx.x >> 6;        // tensor (0 = s, 1 = n)
    int d = blockIdx.x & 63;
    const float* x = t ? xn : xs;
    float s = 0.f, s2 = 0.f;
    for (int idx = threadIdx.x; idx < M_B * KDIM * SDIM; idx += blockDim.x) {
        int m = idx >> 14, i = idx & 16383;
        float v = x[((size_t)(m * D_CH + d) << 14) + i];
        s += v; s2 += v * v;
    }
    __shared__ float rs[256], rq[256];
    rs[threadIdx.x] = s; rq[threadIdx.x] = s2;
    __syncthreads();
    for (int o = 128; o > 0; o >>= 1) {
        if (threadIdx.x < o) { rs[threadIdx.x] += rs[threadIdx.x + o];
                               rq[threadIdx.x] += rq[threadIdx.x + o]; }
        __syncthreads();
    }
    if (threadIdx.x == 0) {
        const float n = (float)(M_B * KDIM * SDIM);
        float mu = rs[0] / n;
        float var = rq[0] / n - mu * mu;
        float rstd = rsqrtf(var + 1e-5f);
        float w = bn_w[t * D_CH + d], b = bn_b[t * D_CH + d];
        scale[t * D_CH + d] = rstd * w;
        shift[t * D_CH + d] = b - mu * rstd * w;
    }
}

// ---------------------------------------------------------------------------
// f32 -> f16 weight conversion
// ---------------------------------------------------------------------------
__global__ __launch_bounds__(256)
void cvt_f16_kernel(const float* __restrict__ src, _Float16* __restrict__ dst, int n) {
    int i = blockIdx.x * blockDim.x + threadIdx.x;
    if (i < n) dst[i] = (_Float16)src[i];
}

// ---------------------------------------------------------------------------
// Normalize + gather into [N*L, 64] f16 per pass layout
// ---------------------------------------------------------------------------
__global__ __launch_bounds__(256)
void pack_xn_kernel(const float* __restrict__ xs, const float* __restrict__ xn,
                    const float* __restrict__ scale, const float* __restrict__ shift,
                    _Float16* __restrict__ Xh, int pass) {
    const int total = 2 * TEN_ELEMS;
    for (int idx = blockIdx.x * blockDim.x + threadIdx.x; idx < total;
         idx += gridDim.x * blockDim.x) {
        int st  = idx >> 21;
        int rem = idx & (TEN_ELEMS - 1);          // flat [m][d][k][s]
        int m = rem >> 20;
        int d = (rem >> 14) & 63;
        int k = (rem >> 7) & 127;
        int s = rem & 127;
        const float* x = st ? xn : xs;
        float v = x[rem] * scale[st * D_CH + d] + shift[st * D_CH + d];
        int nsq, l;
        if (pass == 0) { nsq = m * SDIM + s; l = k; }
        else           { nsq = m * KDIM + k; l = s; }
        Xh[(size_t)st * ((size_t)NL * D_CH) + ((size_t)(nsq * LSEQ + l)) * D_CH + d] =
            (_Float16)v;
    }
}

// ---------------------------------------------------------------------------
// Fused attention kernel: one block per (sequence n, head h, stream st)
//  projections (WMMA) -> scores (WMMA) -> dual softmax -> att@V (WMMA)
// ---------------------------------------------------------------------------
__device__ __forceinline__
void gemm_proj(const _Float16* __restrict__ Xlds, const _Float16* __restrict__ Wg,
               const float* __restrict__ bg, _Float16* __restrict__ Dst,
               int wave, int ln, int hi) {
    int row0 = wave * 16;
#pragma unroll
    for (int nt = 0; nt < 4; ++nt) {
        v8f acc = {};
#pragma unroll
        for (int kt = 0; kt < 2; ++kt) {
            v16h a = load_A(Xlds, D_CH, row0, kt * 32, ln, hi);
            v16h b = load_B_nk(Wg, D_CH, nt * 16, kt * 32, ln, hi);
            acc = __builtin_amdgcn_wmma_f32_16x16x32_f16(false, a, false, b,
                                                         (short)0, acc, false, false);
        }
        int col = nt * 16 + ln;
        float bias = bg[col];
#pragma unroll
        for (int r = 0; r < 8; ++r) {
            float v = acc[r] + bias;
            Dst[(row0 + hi * 8 + r) * D_CH + col] = (_Float16)(v > 0.f ? v : 0.f);
        }
    }
}

__device__ __forceinline__
void gemm_scores(const _Float16* __restrict__ Q, const _Float16* __restrict__ Kl,
                 float* __restrict__ Sdst, int wave, int ln, int hi) {
    int row0 = wave * 16;
#pragma unroll
    for (int nt = 0; nt < 8; ++nt) {
        v8f acc = {};
#pragma unroll
        for (int kt = 0; kt < 2; ++kt) {
            v16h a = load_A(Q, D_CH, row0, kt * 32, ln, hi);
            v16h b = load_B_nk(Kl, D_CH, nt * 16, kt * 32, ln, hi);
            acc = __builtin_amdgcn_wmma_f32_16x16x32_f16(false, a, false, b,
                                                         (short)0, acc, false, false);
        }
        int col = nt * 16 + ln;
#pragma unroll
        for (int r = 0; r < 8; ++r)
            Sdst[(row0 + hi * 8 + r) * LSEQ + col] = acc[r];
    }
}

__global__ __launch_bounds__(256)
void attn_kernel(const _Float16* __restrict__ Xh, const _Float16* __restrict__ Whin,
                 const float* __restrict__ b_in, _Float16* __restrict__ Yh) {
    extern __shared__ char smem[];
    _Float16* Xs  = (_Float16*)smem;          // [128][64] self (reused as Att)
    _Float16* Xo  = Xs + 8192;                // [128][64] other stream
    _Float16* Qs  = Xo + 8192;
    _Float16* Qc  = Qs + 8192;
    _Float16* Km  = Qc + 8192;
    _Float16* Vm  = Km + 8192;
    float*    Ss  = (float*)(Vm + 8192);      // [128][128]
    float*    Sc  = Ss + LSEQ * LSEQ;
    _Float16* Att = Xs;                       // alias: 128*128 halves == Xs+Xo

    const int n  = blockIdx.x;
    const int h  = blockIdx.y;
    const int st = blockIdx.z;
    const int wave = threadIdx.x >> 5;
    const int lane = threadIdx.x & 31;
    const int ln = lane & 15, hi = (lane >> 4) & 1;

    const size_t strideX = (size_t)NL * D_CH;
    const _Float16* x_self = Xh + (size_t)st       * strideX + (size_t)n * (LSEQ * D_CH);
    const _Float16* x_oth  = Xh + (size_t)(1 - st) * strideX + (size_t)n * (LSEQ * D_CH);

    // Stage X tiles into LDS (16 KB each, uint4 vectorized)
    {
        const uint4* ps = (const uint4*)x_self;
        const uint4* po = (const uint4*)x_oth;
        uint4* ds = (uint4*)Xs;
        uint4* do_ = (uint4*)Xo;
        for (int i = threadIdx.x; i < 1024; i += 256) { ds[i] = ps[i]; do_[i] = po[i]; }
    }
    __syncthreads();

    // Projections: q_self(own), q_cross(other stream's q), k(own), v(own)
    const int p_qs = st * 4 + 0;
    const int p_qc = (1 - st) * 4 + 1;
    const int p_k  = st * 4 + 2;
    const int p_v  = st * 4 + 3;
    __builtin_prefetch(Whin + ((size_t)p_qs * 256 + h * 64) * D_CH, 0, 3);
    gemm_proj(Xs, Whin + ((size_t)p_qs * 256 + h * 64) * D_CH, b_in + p_qs * 256 + h * 64, Qs, wave, ln, hi);
    gemm_proj(Xo, Whin + ((size_t)p_qc * 256 + h * 64) * D_CH, b_in + p_qc * 256 + h * 64, Qc, wave, ln, hi);
    gemm_proj(Xs, Whin + ((size_t)p_k  * 256 + h * 64) * D_CH, b_in + p_k  * 256 + h * 64, Km, wave, ln, hi);
    gemm_proj(Xs, Whin + ((size_t)p_v  * 256 + h * 64) * D_CH, b_in + p_v  * 256 + h * 64, Vm, wave, ln, hi);
    __syncthreads();

    // Scores (raw dot products; scaled inside softmax)
    gemm_scores(Qs, Km, Ss, wave, ln, hi);
    gemm_scores(Qc, Km, Sc, wave, ln, hi);
    __syncthreads();

    // att = 0.5 * (softmax(s_self) + softmax(-s_cross)) per row
    if (threadIdx.x < LSEQ) {
        const int l = threadIdx.x;
        const float scl = 0.125f;               // 1/sqrt(dh=64)
        float ms = -3.0e38f, mc = -3.0e38f;
        for (int m2 = 0; m2 < LSEQ; ++m2) {
            float a =  Ss[l * LSEQ + m2] * scl; ms = a > ms ? a : ms;
            float c = -Sc[l * LSEQ + m2] * scl; mc = c > mc ? c : mc;
        }
        float sums = 0.f, sumc = 0.f;
        for (int m2 = 0; m2 < LSEQ; ++m2) {
            sums += __expf( Ss[l * LSEQ + m2] * scl - ms);
            sumc += __expf(-Sc[l * LSEQ + m2] * scl - mc);
        }
        float is = 0.5f / sums, ic = 0.5f / sumc;
        for (int m2 = 0; m2 < LSEQ; ++m2) {
            float es = __expf( Ss[l * LSEQ + m2] * scl - ms) * is;
            float ec = __expf(-Sc[l * LSEQ + m2] * scl - mc) * ic;
            Att[l * LSEQ + m2] = (_Float16)(es + ec);   // overwrites Xs/Xo (done)
        }
    }
    __syncthreads();

    // Y = att @ V  -> global Yh[st][n][l][h*64 + col]
    {
        int row0 = wave * 16;
        _Float16* Yg = Yh + (size_t)st * ((size_t)NL * 256);
#pragma unroll
        for (int nt = 0; nt < 4; ++nt) {
            v8f acc = {};
#pragma unroll
            for (int kt = 0; kt < 4; ++kt) {
                v16h a = load_A(Att, LSEQ, row0, kt * 32, ln, hi);
                v16h b = load_B_kn(Vm, D_CH, kt * 32, nt * 16, ln, hi);
                acc = __builtin_amdgcn_wmma_f32_16x16x32_f16(false, a, false, b,
                                                             (short)0, acc, false, false);
            }
            int col = nt * 16 + ln;
#pragma unroll
            for (int r = 0; r < 8; ++r) {
                int row = row0 + hi * 8 + r;
                Yg[((size_t)n * LSEQ + row) * 256 + h * 64 + col] = (_Float16)acc[r];
            }
        }
    }
}

// ---------------------------------------------------------------------------
// Output projection: Z[m,d,k,s] = relu(Y[n,l,:256] . W_out[d,:256] + b)
// ---------------------------------------------------------------------------
__global__ __launch_bounds__(256)
void outproj_kernel(const _Float16* __restrict__ Yh, const _Float16* __restrict__ Whout,
                    const float* __restrict__ b_out, float* __restrict__ Z, int pass) {
    const int st = blockIdx.y;
    const int R0 = blockIdx.x * 128;
    const int wave = threadIdx.x >> 5;
    const int lane = threadIdx.x & 31;
    const int ln = lane & 15, hi = (lane >> 4) & 1;
    const _Float16* Y = Yh + (size_t)st * ((size_t)NL * 256);
    const _Float16* W = Whout + (size_t)st * (D_CH * 256);
    const int row0 = R0 + wave * 16;
#pragma unroll
    for (int nt = 0; nt < 4; ++nt) {
        v8f acc = {};
#pragma unroll
        for (int kt = 0; kt < 8; ++kt) {
            v16h a, b;
#pragma unroll
            for (int e = 0; e < 16; ++e)
                a[e] = Y[(size_t)(row0 + ln) * 256 + kt * 32 + a_kidx(e, hi)];
#pragma unroll
            for (int e = 0; e < 16; ++e)
                b[e] = W[(nt * 16 + ln) * 256 + kt * 32 + hi * 16 + e];
            acc = __builtin_amdgcn_wmma_f32_16x16x32_f16(false, a, false, b,
                                                         (short)0, acc, false, false);
        }
        int dch = nt * 16 + ln;
        float bias = b_out[st * D_CH + dch];
#pragma unroll
        for (int r = 0; r < 8; ++r) {
            int row = row0 + hi * 8 + r;
            int nsq = row >> 7, l = row & 127;
            int m, kk, ss;
            if (pass == 0) { m = nsq >> 7; ss = nsq & 127; kk = l; }
            else           { m = nsq >> 7; kk = nsq & 127; ss = l; }
            float v = acc[r] + bias;
            v = v > 0.f ? v : 0.f;
            Z[(size_t)st * TEN_ELEMS + (((size_t)(m * D_CH + dch)) << 14) + kk * SDIM + ss] = v;
        }
    }
}

// ---------------------------------------------------------------------------
// GroupNorm(1, D) statistics per (stream, sample)
// ---------------------------------------------------------------------------
__global__ __launch_bounds__(256)
void gn_stats_kernel(const float* __restrict__ Z, float* __restrict__ gmu,
                     float* __restrict__ grs) {
    const int st = blockIdx.x >> 1, m = blockIdx.x & 1;
    const float* z = Z + (size_t)st * TEN_ELEMS + ((size_t)m << 20);
    float s = 0.f, s2 = 0.f;
    for (int i = threadIdx.x; i < (1 << 20); i += blockDim.x) {
        float v = z[i]; s += v; s2 += v * v;
    }
    __shared__ float rs[256], rq[256];
    rs[threadIdx.x] = s; rq[threadIdx.x] = s2;
    __syncthreads();
    for (int o = 128; o > 0; o >>= 1) {
        if (threadIdx.x < o) { rs[threadIdx.x] += rs[threadIdx.x + o];
                               rq[threadIdx.x] += rq[threadIdx.x + o]; }
        __syncthreads();
    }
    if (threadIdx.x == 0) {
        const float n = 1048576.f;
        float mu = rs[0] / n;
        float var = rq[0] / n - mu * mu;
        gmu[blockIdx.x] = mu;
        grs[blockIdx.x] = rsqrtf(var + 1e-8f);
    }
}

// ---------------------------------------------------------------------------
// GroupNorm apply + residual; writes running state into d_out
// ---------------------------------------------------------------------------
__global__ __launch_bounds__(256)
void gn_apply_kernel(const float* __restrict__ Z, const float* __restrict__ gmu,
                     const float* __restrict__ grs, const float* __restrict__ gn_w,
                     const float* __restrict__ gn_b, const float* __restrict__ prev_s,
                     const float* __restrict__ prev_n, float* __restrict__ out, int pass) {
    const int total = 2 * TEN_ELEMS;
    for (int idx = blockIdx.x * blockDim.x + threadIdx.x; idx < total;
         idx += gridDim.x * blockDim.x) {
        int st  = idx >> 21;
        int rem = idx & (TEN_ELEMS - 1);
        int m = rem >> 20;
        int d = (rem >> 14) & 63;
        int gidx = pass * 2 + st;
        float pv = (st ? prev_n : prev_s)[rem];
        float z = Z[idx];
        float v = (z - gmu[st * 2 + m]) * grs[st * 2 + m] * gn_w[gidx * D_CH + d]
                  + gn_b[gidx * D_CH + d] + pv;
        out[idx] = v;
    }
}

// ---------------------------------------------------------------------------
extern "C" void kernel_launch(void* const* d_in, const int* in_sizes, int n_in,
                              void* d_out, int out_size, void* d_ws, size_t ws_size,
                              hipStream_t stream) {
    const float* x_s        = (const float*)d_in[0];
    const float* x_n        = (const float*)d_in[1];
    const float* W_in[2]    = { (const float*)d_in[2],  (const float*)d_in[8]  };
    const float* b_in[2]    = { (const float*)d_in[3],  (const float*)d_in[9]  };
    const float* W_out[2]   = { (const float*)d_in[4],  (const float*)d_in[10] };
    const float* b_out[2]   = { (const float*)d_in[5],  (const float*)d_in[11] };
    const float* bn_w[2]    = { (const float*)d_in[6],  (const float*)d_in[12] };
    const float* bn_b[2]    = { (const float*)d_in[7],  (const float*)d_in[13] };
    const float* gn_w       = (const float*)d_in[14];
    const float* gn_b       = (const float*)d_in[15];
    float* out = (float*)d_out;                // [x_s | x_n], TEN_ELEMS each

    // Workspace carve (256B aligned)
    char* base = (char*)d_ws;
    size_t off = 0;
    auto carve = [&](size_t bytes) -> char* {
        char* p = base + off;
        off = (off + bytes + 255) & ~(size_t)255;
        return p;
    };
    float*    bn_scale = (float*)carve(2 * D_CH * sizeof(float));
    float*    bn_shift = (float*)carve(2 * D_CH * sizeof(float));
    float*    gmu      = (float*)carve(4 * sizeof(float));
    float*    grs      = (float*)carve(4 * sizeof(float));
    _Float16* Whin     = (_Float16*)carve((size_t)8 * 256 * 64 * sizeof(_Float16));
    _Float16* Whout    = (_Float16*)carve((size_t)2 * 64 * 256 * sizeof(_Float16));
    _Float16* Xh       = (_Float16*)carve((size_t)2 * NL * D_CH * sizeof(_Float16));
    _Float16* Yh       = (_Float16*)carve((size_t)2 * NL * 256 * sizeof(_Float16));
    float*    Z        = (float*)carve((size_t)2 * TEN_ELEMS * sizeof(float));
    (void)ws_size; (void)in_sizes; (void)n_in; (void)out_size;

    const size_t attn_lds = 6 * 8192 * sizeof(_Float16)       // X/Q/K/V tiles
                          + 2 * LSEQ * LSEQ * sizeof(float);  // Ss, Sc  (224 KB)

    for (int pass = 0; pass < 2; ++pass) {
        const float* xs_src = pass ? out               : x_s;
        const float* xn_src = pass ? out + TEN_ELEMS   : x_n;

        cvt_f16_kernel<<<(8 * 256 * 64 + 255) / 256, 256, 0, stream>>>(W_in[pass], Whin, 8 * 256 * 64);
        cvt_f16_kernel<<<(2 * 64 * 256 + 255) / 256, 256, 0, stream>>>(W_out[pass], Whout, 2 * 64 * 256);

        bn_stats_kernel<<<2 * D_CH, 256, 0, stream>>>(xs_src, xn_src, bn_w[pass], bn_b[pass],
                                                      bn_scale, bn_shift);

        pack_xn_kernel<<<4096, 256, 0, stream>>>(xs_src, xn_src, bn_scale, bn_shift, Xh, pass);

        dim3 agrid(NSEQ, 4, 2);   // (sequence, head, stream)
        attn_kernel<<<agrid, 256, attn_lds, stream>>>(Xh, Whin, b_in[pass], Yh);

        outproj_kernel<<<dim3(NL / 128, 2), 256, 0, stream>>>(Yh, Whout, b_out[pass], Z, pass);

        gn_stats_kernel<<<4, 256, 0, stream>>>(Z, gmu, grs);

        gn_apply_kernel<<<4096, 256, 0, stream>>>(Z, gmu, grs, gn_w, gn_b,
                                                  xs_src, xn_src, out, pass);
    }
}
